// FuzzyCausalConv1d_9818295238774
// MI455X (gfx1250) — matrix operand order repaired and checked by reading, not compiled, not run on hardware
//
#include <hip/hip_runtime.h>

// ---------------- types ----------------
typedef __attribute__((ext_vector_type(16))) __bf16  v16bf;
typedef __attribute__((ext_vector_type(8)))  float   v8f;
typedef __attribute__((ext_vector_type(4)))  unsigned v4u;
typedef __attribute__((ext_vector_type(8)))  unsigned v8u;

#define B_     16
#define CIN    256
#define COUT   256
#define KW     3
#define L_     4096
#define NR     5
#define CKT    (CIN * KW)          // 768, K reordered kw-major: K = kw*256 + c
#define TILE_L 128                 // 8 position subtiles of 16; 2 per wave
#define TILE_CO 32                 // 2 cout subtiles of 16
#define KCH    32                  // WMMA K per step
#define NKSTEP (CKT / KCH)         // 24 (chunks 0..7: kw=0, 8..15: kw=1, 16..23: kw=2)

__device__ __forceinline__ unsigned short f32_to_bf16(float f) {
    unsigned u = __float_as_uint(f);
    u += 0x7FFFu + ((u >> 16) & 1u);   // round-to-nearest-even
    return (unsigned short)(u >> 16);
}

// ---------------- kernel 0a: x fp32 -> bf16 ----------------
__global__ void cvt_f32_bf16(const float* __restrict__ src,
                             unsigned short* __restrict__ dst, int n) {
    int i = blockIdx.x * blockDim.x + threadIdx.x;
    if (i < n) dst[i] = f32_to_bf16(src[i]);
}

// ---------------- kernel 0b: weights fp32 -> bf16, K permuted kw-major ----
// src idx = ((n*COUT+o)*CIN + c)*KW + kw  ->  dst[(n*COUT+o)*CKT + kw*CIN + c]
__global__ void cvt_weights(const float* __restrict__ src,
                            unsigned short* __restrict__ dst) {
    int i = blockIdx.x * blockDim.x + threadIdx.x;
    if (i < NR * COUT * CIN * KW) {
        int kw  = i % KW;
        int t   = i / KW;
        int c   = t % CIN;
        int row = t / CIN;
        dst[(size_t)row * CKT + kw * CIN + c] = f32_to_bf16(src[i]);
    }
}

// ---------------- kernel 1: controller (fp32 dot + softmax) ----------------
__global__ __launch_bounds__(256)
void controller_kernel(const float* __restrict__ x,       // (B,CIN,L)
                       const float* __restrict__ ctrl_w,  // (CKT,NR) natural order
                       const float* __restrict__ ctrl_b,  // (NR)
                       float* __restrict__ fr) {          // (B*L, NR)
    int idx = blockIdx.x * blockDim.x + threadIdx.x;      // over B*L
    int b = idx >> 12;
    int l = idx & (L_ - 1);
    float lg[NR];
#pragma unroll
    for (int n = 0; n < NR; ++n) lg[n] = ctrl_b[n];

    const float* xr = x + (size_t)b * CIN * L_;
    for (int c = 0; c < CIN; ++c) {
        const float* xc = xr + (size_t)c * L_;
#pragma unroll
        for (int k = 0; k < KW; ++k) {
            int ll = l + k - (KW - 1);
            float xv = (ll >= 0) ? xc[ll] : 0.0f;
            const float* w = ctrl_w + (c * KW + k) * NR;
#pragma unroll
            for (int n = 0; n < NR; ++n) lg[n] = __fmaf_rn(xv, w[n], lg[n]);
        }
    }
    float mx = lg[0];
#pragma unroll
    for (int n = 1; n < NR; ++n) mx = fmaxf(mx, lg[n]);
    float s = 0.0f;
#pragma unroll
    for (int n = 0; n < NR; ++n) { lg[n] = __expf(lg[n] - mx); s += lg[n]; }
    float inv = 1.0f / s;
    float* o = fr + (size_t)idx * NR;
#pragma unroll
    for (int n = 0; n < NR; ++n) o[n] = lg[n] * inv;
}

// ---------------- TDM: 3D tile load Wb -> LDS (CDNA5 ISA) ------------
// Tile: dim0 = 32 K (contiguous), dim1 = 32 cout rows (stride CKT),
//       dim2 = NR rules (stride COUT*CKT). data_size = 2 bytes.
// LDS fill order X,Y,Z  ==  ldA[rule][m][k] contiguous (10 KB).
__device__ __forceinline__ void tdm_load_A(const unsigned short* gptr,
                                           unsigned lds_addr) {
    unsigned long long ga = (unsigned long long)(size_t)gptr;
    v4u g0;
    g0.x = 1u;                                        // count=1 (valid), user mode
    g0.y = lds_addr;                                  // [63:32] lds_addr
    g0.z = (unsigned)(ga & 0xFFFFFFFFull);            // [95:64] global_addr lo
    g0.w = ((unsigned)(ga >> 32) & 0x01FFFFFFu) | (2u << 30);  // addr hi | type=2

    const unsigned dim0 = CKT;                        // tensor_dim0
    const unsigned dim1 = COUT;                       // tensor_dim1
    const unsigned td0 = KCH, td1 = 32, td2 = NR;     // tile dims
    const unsigned long long s0 = CKT;                // dim0 stride (elems)
    const unsigned long long s1 = (unsigned long long)COUT * CKT;  // dim1 stride
    v8u g1;
    g1.s0 = 0x00010000u;                              // data_size=1 (2 bytes)
    g1.s1 = (dim0 & 0xFFFFu) << 16;                   // abar_addr=0 | dim0 lo16
    g1.s2 = (dim0 >> 16) | ((dim1 & 0xFFFFu) << 16);  // dim0 hi16 | dim1 lo16
    g1.s3 = (dim1 >> 16) | (td0 << 16);               // dim1 hi16 | tile_dim0
    g1.s4 = td1 | (td2 << 16);                        // tile_dim1 | tile_dim2
    g1.s5 = (unsigned)(s0 & 0xFFFFFFFFull);           // stride0 lo32
    g1.s6 = (unsigned)(s0 >> 32) | ((unsigned)(s1 & 0xFFFFull) << 16); // s0 hi | s1 lo16
    g1.s7 = (unsigned)(s1 >> 16);                     // stride1 [47:16]

    v4u g2;
    g2.x = NR;                                        // tensor_dim2
    g2.y = 0u;                                        // tensor_dim3 (unused)
    g2.z = 0u;                                        // tensor_dim2_stride (unused, 3D)
    g2.w = 0u;                                        // stride hi | tile_dim3=0
    v4u g3 = (v4u){0u, 0u, 0u, 0u};

    asm volatile("tensor_load_to_lds %0, %1, %2, %3"
                 :: "s"(g0), "s"(g1), "s"(g2), "s"(g3)
                 : "memory");
}

// ---------------- kernel 2: fused 5-expert conv GEMM + mix ----------------
__global__ __launch_bounds__(256)
void fuzzy_main(const unsigned short* __restrict__ xb,   // bf16 (B,CIN,L)
                const unsigned short* __restrict__ Wb,   // bf16 (NR*COUT, CKT) kw-major K
                const float* __restrict__ bias,          // (NR, COUT)
                const float* __restrict__ fr,            // (B*L, NR)
                float* __restrict__ out) {               // (B, COUT, L)
    __shared__ __align__(16) unsigned short ldA[2][NR][32][KCH];   // 2 x 10 KB
    __shared__ __align__(16) unsigned short ldB[2][TILE_L][KCH];   // 2 x 8 KB, pos-major

    const int tid  = threadIdx.x;
    const int lane = tid & 31;
    const int wave = tid >> 5;
    const int wm   = wave & 1;        // cout subtile (0..1)
    const int wn   = wave >> 1;       // owns positions [wn*32, wn*32+32)
    const int lb   = blockIdx.x * TILE_L;
    const int cob  = blockIdx.y * TILE_CO;
    const int b    = blockIdx.z;

    const int half = lane >> 4;
    const int lm   = lane & 15;

    const int kRow = tid >> 3;        // B stager: K row 0..31
    const int p0   = (tid & 7) * 16;  // 16 consecutive positions

    v8f acc0[NR], acc1[NR];
#pragma unroll
    for (int n = 0; n < NR; ++n) {
        acc0[n] = (v8f){0.f,0.f,0.f,0.f,0.f,0.f,0.f,0.f};
        acc1[n] = (v8f){0.f,0.f,0.f,0.f,0.f,0.f,0.f,0.f};
    }

    // ---- prologue: fill buffer 0 ----
    if (tid == 0)
        tdm_load_A(Wb + (size_t)cob * CKT + 0 * KCH,
                   (unsigned)(size_t)&ldA[0][0][0][0]);
    {
        int c = kRow;                 // chunk 0: kw=0, cbase=0
        int s = 0 - (KW - 1);
        const unsigned short* xr = xb + (size_t)(b * CIN + c) * L_;
#pragma unroll
        for (int i = 0; i < 16; ++i) {
            int l = lb + p0 + i + s;
            ldB[0][p0 + i][kRow] = (l >= 0) ? xr[l] : (unsigned short)0;
        }
    }
    __builtin_amdgcn_s_wait_tensorcnt(0);
    __syncthreads();

    for (int kk = 0; kk < NKSTEP; ++kk) {
        const int cur = kk & 1;
        const int nxt = cur ^ 1;
        const int kn  = kk + 1;
        const bool haveNext = (kn < NKSTEP);

        // ---- issue TDM DMA of next A tile (one wave, EXEC-independent) ----
        if (haveNext && tid == 0)
            tdm_load_A(Wb + (size_t)cob * CKT + (size_t)kn * KCH,
                       (unsigned)(size_t)&ldA[nxt][0][0][0]);

        // ---- fetch next B chunk into registers (overlaps with WMMA) ----
        unsigned short breg[16];
        if (haveNext) {
            int c = ((kn & 7) << 5) + kRow;          // cbase + row (kw-major K)
            int s = (kn >> 3) - (KW - 1);            // uniform shift for this chunk
            const unsigned short* xr = xb + (size_t)(b * CIN + c) * L_;
#pragma unroll
            for (int i = 0; i < 16; ++i) {
                int l = lb + p0 + i + s;
                breg[i] = (l >= 0) ? xr[l] : (unsigned short)0;
            }
        }

        // ---- fragments + 10 WMMAs on current buffers ----
        union { v16bf v; unsigned int u[8]; } bfr0, bfr1;
#pragma unroll
        for (int j = 0; j < 8; ++j) {
            int kob = (j << 1) + (half << 4);        // B: n=lm, K=2j (+16 hi half)
            bfr0.u[j] = *(const unsigned int*)&ldB[cur][wn * 32 + lm][kob];
            bfr1.u[j] = *(const unsigned int*)&ldB[cur][wn * 32 + 16 + lm][kob];
        }
#pragma unroll
        for (int n = 0; n < NR; ++n) {
            union { v16bf v; unsigned int u[8]; } afr;
#pragma unroll
            for (int j = 0; j < 8; ++j) {
                int koa = (j << 1) + (half << 4);  // A layout
                afr.u[j] = *(const unsigned int*)&ldA[cur][n][wm * 16 + lm][koa];
            }
            acc0[n] = __builtin_amdgcn_wmma_f32_16x16x32_bf16(
                          false, afr.v, false, bfr0.v, (short)0, acc0[n], false, false);
            acc1[n] = __builtin_amdgcn_wmma_f32_16x16x32_bf16(
                          false, afr.v, false, bfr1.v, (short)0, acc1[n], false, false);
        }

        // ---- commit staged B chunk to next buffer ----
        if (haveNext) {
#pragma unroll
            for (int i = 0; i < 16; ++i)
                ldB[nxt][p0 + i][kRow] = breg[i];
        }

        __builtin_amdgcn_s_wait_tensorcnt(0);   // A[nxt] DMA complete (wave 0)
        __syncthreads();                        // publish A[nxt], B[nxt]
    }

    // ---- epilogue: out[b,o,l] = sum_n fr[b,l,n] * (acc_n + bias[n,o]) ----
#pragma unroll
    for (int t = 0; t < 2; ++t) {
        const v8f* acc = t ? acc1 : acc0;
        const int lpos = lb + wn * 32 + t * 16 + lm;     // N = lm per lane
        const float* frp = fr + (size_t)(b * L_ + lpos) * NR;
        float f0 = frp[0], f1 = frp[1], f2 = frp[2], f3 = frp[3], f4 = frp[4];
#pragma unroll
        for (int v = 0; v < 8; ++v) {
            int o = cob + wm * 16 + v + half * 8;        // M = v + 8*half
            float r = f0 * (acc[0][v] + bias[0 * COUT + o])
                    + f1 * (acc[1][v] + bias[1 * COUT + o])
                    + f2 * (acc[2][v] + bias[2 * COUT + o])
                    + f3 * (acc[3][v] + bias[3 * COUT + o])
                    + f4 * (acc[4][v] + bias[4 * COUT + o]);
            out[(size_t)(b * COUT + o) * L_ + lpos] = r;
        }
    }
}

// ---------------- launcher ----------------
extern "C" void kernel_launch(void* const* d_in, const int* in_sizes, int n_in,
                              void* d_out, int out_size, void* d_ws, size_t ws_size,
                              hipStream_t stream) {
    const float* x            = (const float*)d_in[0];   // (16,256,4096)
    const float* base_kernels = (const float*)d_in[1];   // (5,256,256,3)
    const float* base_biases  = (const float*)d_in[2];   // (5,256)
    const float* ctrl_w       = (const float*)d_in[3];   // (768,5)
    const float* ctrl_b       = (const float*)d_in[4];   // (5)
    float* out = (float*)d_out;

    char* ws = (char*)d_ws;
    const size_t NX = (size_t)B_ * CIN * L_;             // 16.7M elems
    const size_t NW = (size_t)NR * COUT * CKT;           // 983k elems
    unsigned short* xb = (unsigned short*)ws;                         // 32 MB
    unsigned short* Wb = (unsigned short*)(ws + NX * 2);              // ~2 MB
    float*          fr = (float*)(ws + NX * 2 + ((NW * 2 + 255) & ~(size_t)255));

    cvt_f32_bf16<<<(int)((NX + 255) / 256), 256, 0, stream>>>(x, xb, (int)NX);
    cvt_weights<<<(int)((NW + 255) / 256), 256, 0, stream>>>(base_kernels, Wb);
    controller_kernel<<<(B_ * L_) / 256, 256, 0, stream>>>(x, ctrl_w, ctrl_b, fr);

    dim3 grid(L_ / TILE_L, COUT / TILE_CO, B_);          // 32 x 8 x 16
    fuzzy_main<<<grid, 256, 0, stream>>>(xb, Wb, base_biases, fr, out);
}
